// PPONetwork_9182640079188
// MI455X (gfx1250) — matrix operand order repaired and checked by reading, not compiled, not run on hardware
//
#include <hip/hip_runtime.h>
#include <hip/hip_bf16.h>

// ---------------------------------------------------------------------------
// PPO network inference, fused for gfx1250 (MI455X).
// Compute-bound (~83 GFLOP vs ~60 MB of HBM traffic) -> all GEMMs run through
// v_wmma_f32_16x16x32_f16. Weights are pre-packed into WMMA B-fragment layout
// in d_ws (each lane's 16 f16 B elements = one contiguous 32B read).
// This revision: M=32 per wave (two A tiles share every B fragment -> half the
// L2 B traffic, 2 independent WMMAs per load) + manual double-buffering of B
// fragments and next-kt A fragments so loads overlap matrix math instead of
// draining loadcnt to zero before every WMMA.
// ---------------------------------------------------------------------------

typedef __attribute__((ext_vector_type(16))) _Float16 v16h;
typedef __attribute__((ext_vector_type(8)))  float    v8f;

#define XSTR   576        // X activation stride: 512 base + 64 card-emb cols
#define FRAGE  512        // f16 elements per 16x16x32 B fragment (32 lanes * 16)

// ws fragment offsets (in fragments). Frag order within a layer: kt-major, nt-minor.
#define OFF_HAND  0       // 256x32   : kt=1,  nt=16 -> 16 frags
#define OFF_STATE 16      // 256x133  : kt=5,  nt=16 -> 80 frags
#define OFF_ST0   96      // 256x512  : kt=16, nt=16 -> 256
#define OFF_ST1   352     // 256x528  : kt=17, nt=16 -> 272
#define OFF_ST2   624     // 256x544  : kt=17, nt=16 -> 272
#define OFF_ST3   896     // 256x560  : kt=18, nt=16 -> 288
#define OFF_HEAD  1184    // 4 x (32x256 : kt=8, nt=2) -> 64
#define OFF_COMB  1248    // 20(->32)x512 : kt=16, nt=2 -> 32
#define TOTAL_FRAGS 1280  // 1280 * 1KB = 1.25 MB of d_ws

struct PackParams {
    const float *hand_W, *state_W, *st0, *st1, *st2, *st3;
    const float *h0, *h1, *h2, *h3;
    const float *bid_W, *ann_W, *val_W;
    _Float16* ws;
};

__global__ __launch_bounds__(256) void ppo_pack_weights(PackParams pp) {
    int fi = blockIdx.x;
    const float* W = nullptr; int N = 0, K = 0, nt = 0, kt = 0, comb = 0;
    if (fi < 16)        { W = pp.hand_W;  N = 256; K = 32;  int l = fi;       kt = l / 16; nt = l % 16; }
    else if (fi < 96)   { W = pp.state_W; N = 256; K = 133; int l = fi - 16;  kt = l / 16; nt = l % 16; }
    else if (fi < 352)  { W = pp.st0;     N = 256; K = 512; int l = fi - 96;  kt = l / 16; nt = l % 16; }
    else if (fi < 624)  { W = pp.st1;     N = 256; K = 528; int l = fi - 352; kt = l / 16; nt = l % 16; }
    else if (fi < 896)  { W = pp.st2;     N = 256; K = 544; int l = fi - 624; kt = l / 16; nt = l % 16; }
    else if (fi < 1184) { W = pp.st3;     N = 256; K = 560; int l = fi - 896; kt = l / 16; nt = l % 16; }
    else if (fi < 1248) {
        int l = fi - 1184; int s = l / 16; l %= 16;
        const float* hw[4] = {pp.h0, pp.h1, pp.h2, pp.h3};
        W = hw[s]; N = 32; K = 256; kt = l / 2; nt = l % 2;
    } else {
        int l = fi - 1248; comb = 1; N = 32; K = 512; kt = l / 2; nt = l % 2;
    }
    // B-fragment layout (16x16x32 f16): lane 0-15 -> N=0..15 holding K {0..7,16..23};
    // lanes 16-31 same N, K {8..15,24..31}.
    for (int e = threadIdx.x; e < FRAGE; e += 256) {
        int lane = e >> 4, i = e & 15;
        int n = nt * 16 + (lane & 15);
        int k = kt * 32 + ((lane >= 16) ? 8 : 0) + ((i < 8) ? i : 8 + i);
        float v = 0.f;
        if (!comb) {
            if (n < N && k < K) v = W[(size_t)n * K + k];
        } else if (k < 512) {
            if (n < 9)       v = pp.bid_W[(size_t)n * 512 + k];
            else if (n < 19) v = pp.ann_W[(size_t)(n - 9) * 512 + k];
            else if (n == 19) v = pp.val_W[k];
        }
        pp.ws[(size_t)fi * FRAGE + e] = (_Float16)v;
    }
}

struct NetParams {
    const float* hand; const float* prob; const int* tables; const float* trumps;
    const float* emb;
    const float* hand_b; const float* state_b;
    const float* stage_b0; const float* stage_b1; const float* stage_b2; const float* stage_b3;
    const float* head_b0;  const float* head_b1;  const float* head_b2;  const float* head_b3;
    const float* bid_b; const float* ann_b; const float* val_b;
    const _Float16* ws;
    float* out;
    int nB;
};

__device__ inline v8f vzero8() {
    v8f z;
#pragma unroll
    for (int i = 0; i < 8; ++i) z[i] = 0.f;
    return z;
}

__device__ inline v8f wmma_f16(v16h a, v16h b, v8f c) {
    return __builtin_amdgcn_wmma_f32_16x16x32_f16(false, a, false, b, (short)0, c, false, false);
}

// A-fragment 16x32 f16 layout: lanes 0-15 row M=lane hold K {0..7,16..23} in
// element pairs; lanes 16-31 same rows hold K {8..15,24..31}.
__device__ inline v16h load_a_lds(const _Float16* Xrow, int kt, int hi) {
    v16h a;
#pragma unroll
    for (int j = 0; j < 8; ++j) {
        int k0 = kt * 32 + ((j & 3) << 1) + ((j >= 4) ? 16 : 0) + (hi ? 8 : 0);
        a[2 * j]     = Xrow[k0];
        a[2 * j + 1] = Xrow[k0 + 1];
    }
    return a;
}

// bias + LayerNorm + ReLU on one 16-row C tile, f16 result to LDS.
__device__ inline void ln_store(v8f acc[16], const float* bias, _Float16* dst, int dstride,
                                int rowoff, int lane) {
    const int m = lane & 15, hi = lane >> 4;
    float s[8], s2[8];
#pragma unroll
    for (int r = 0; r < 8; ++r) { s[r] = 0.f; s2[r] = 0.f; }
#pragma unroll
    for (int nt = 0; nt < 16; ++nt) {
        float bv = bias[nt * 16 + m];
#pragma unroll
        for (int r = 0; r < 8; ++r) {
            float v = acc[nt][r] + bv;
            acc[nt][r] = v; s[r] += v; s2[r] += v * v;
        }
    }
#pragma unroll
    for (int off = 1; off < 16; off <<= 1) {
#pragma unroll
        for (int r = 0; r < 8; ++r) {
            s[r]  += __shfl_xor(s[r],  off, 32);
            s2[r] += __shfl_xor(s2[r], off, 32);
        }
    }
#pragma unroll
    for (int r = 0; r < 8; ++r) {
        float mean = s[r] * (1.f / 256.f);
        float var  = s2[r] * (1.f / 256.f) - mean * mean;
        float rstd = rsqrtf(var + 1e-5f);
#pragma unroll
        for (int nt = 0; nt < 16; ++nt) {
            float v = (acc[nt][r] - mean) * rstd;
            v = v > 0.f ? v : 0.f;
            dst[(size_t)(rowoff + r + hi * 8) * dstride + nt * 16 + m] = (_Float16)v;
        }
    }
}

// Hidden layer, M=32 (two 16-row A tiles sharing each B fragment), N=256.
// B fragments and next-kt A fragments are double-buffered so loads overlap
// the WMMA chain instead of fully draining loadcnt before each matrix op.
template <typename AF>
__device__ inline void hidden32(AF af, int ksteps, const _Float16* frag, const float* bias,
                                _Float16* dst, int dstride, int lane) {
    v8f acc0[16], acc1[16];
#pragma unroll
    for (int nt = 0; nt < 16; ++nt) { acc0[nt] = vzero8(); acc1[nt] = vzero8(); }

    v16h a0 = af(0, 0), a1 = af(0, 1);
    v16h b = *(const v16h*)(frag + lane * 16);

    for (int kt = 0; kt < ksteps; ++kt) {
        const _Float16* wp = frag + (size_t)kt * 16 * FRAGE + lane * 16;
        v16h ca0 = a0, ca1 = a1;
#pragma unroll
        for (int nt = 0; nt < 16; ++nt) {
            v16h cb = b;
            if (nt < 15)                 b = *(const v16h*)(wp + (nt + 1) * FRAGE);
            else if (kt + 1 < ksteps)    b = *(const v16h*)(wp + 16 * FRAGE);
            if (nt == 0 && kt + 1 < ksteps) { a0 = af(kt + 1, 0); a1 = af(kt + 1, 1); }
            acc0[nt] = wmma_f16(ca0, cb, acc0[nt]);
            acc1[nt] = wmma_f16(ca1, cb, acc1[nt]);
        }
    }
    ln_store(acc0, bias, dst, dstride, 0,  lane);
    ln_store(acc1, bias, dst, dstride, 16, lane);
}

// Small GEMM: 32 rows x 32 cols (2 M-tiles x 2 N-tiles), no LN.
__device__ inline void gemm2x2(const _Float16* Xbase, int xstride, int ksteps,
                               const _Float16* frag, int lane,
                               v8f& c00, v8f& c01, v8f& c10, v8f& c11) {
    const int m = lane & 15, hi = lane >> 4;
    c00 = vzero8(); c01 = vzero8(); c10 = vzero8(); c11 = vzero8();
    for (int kt = 0; kt < ksteps; ++kt) {
        v16h a0 = load_a_lds(Xbase + (size_t)m * xstride,        kt, hi);
        v16h a1 = load_a_lds(Xbase + (size_t)(16 + m) * xstride, kt, hi);
        const _Float16* wp = frag + (size_t)kt * 2 * FRAGE + lane * 16;
        v16h b0 = *(const v16h*)(wp);
        v16h b1 = *(const v16h*)(wp + FRAGE);
        c00 = wmma_f16(a0, b0, c00);
        c10 = wmma_f16(a1, b0, c10);
        c01 = wmma_f16(a0, b1, c01);
        c11 = wmma_f16(a1, b1, c11);
    }
}

__global__ __launch_bounds__(32) void ppo_fused(NetParams P) {
    __shared__ _Float16 X[32][XSTR];   // [hand_f | state_f | 4x card emb]
    __shared__ _Float16 Y[32][256];    // stage output
    __shared__ int      Cnt[32];

    const int lane    = threadIdx.x & 31;
    const int rowBase = blockIdx.x * 32;
    const int m       = lane & 15;
    const int hi      = lane >> 4;

    // ---- card embeddings into X[:,512:576] + valid counts ----
    for (int p = lane; p < 128; p += 32) {
        int row = p >> 2, card = p & 3;
        int b = rowBase + row;
        int t = P.tables[b * 4 + card];
        bool valid = (t >= 0) && (t < 32);
        bool isReg = P.trumps[b * 5] < 0.5f;
        int best = 0; float bv = P.trumps[b * 5 + 1];
        for (int j = 2; j <= 4; ++j) {
            float v = P.trumps[b * 5 + j];
            if (v > bv) { bv = v; best = j - 1; }
        }
        int suit = t >> 3;                     // tables in [0,36)
        bool isTr = isReg && (suit == best) && valid;
        int bidx = valid ? t : 36;
        int sidx = 32 + (valid ? suit : 0);
        for (int d = 0; d < 16; ++d) {
            float v = P.emb[bidx * 16 + d];
            if (isTr) v += P.emb[sidx * 16 + d];
            X[row][512 + card * 16 + d] = (_Float16)v;
        }
    }
    {
        int b = rowBase + lane, c = 0;
        for (int j = 0; j < 4; ++j) {
            int t = P.tables[b * 4 + j];
            c += (t >= 0 && t < 32) ? 1 : 0;
        }
        Cnt[lane] = c;
    }

    // ---- hand block: [32,32] @ [32,256] -> X[:,0:256] ----
    auto handA = [&](int kt, int tile) {
        (void)kt;
        v16h a;
        const float* hrow = P.hand + (size_t)(rowBase + tile * 16 + m) * 32;
#pragma unroll
        for (int j = 0; j < 8; ++j) {
            int k0 = ((j & 3) << 1) + ((j >= 4) ? 16 : 0) + (hi ? 8 : 0);
            a[2 * j]     = (_Float16)hrow[k0];
            a[2 * j + 1] = (_Float16)hrow[k0 + 1];
        }
        return a;
    };
    hidden32(handA, 1, P.ws + (size_t)OFF_HAND * FRAGE, P.hand_b, &X[0][0], XSTR, lane);

    // ---- state block: [32,133pad160] @ [133,256] -> X[:,256:512] ----
    auto stateA = [&](int kt, int tile) {
        v16h a;
        int row = rowBase + tile * 16 + m;
        const float* prow = P.prob   + (size_t)row * 128;
        const float* trow = P.trumps + (size_t)row * 5;
#pragma unroll
        for (int j = 0; j < 8; ++j) {
            int k0 = kt * 32 + ((j & 3) << 1) + ((j >= 4) ? 16 : 0) + (hi ? 8 : 0);
#pragma unroll
            for (int u = 0; u < 2; ++u) {
                int k = k0 + u;
                float v = 0.f;
                if (k < 128)      v = prow[k];
                else if (k < 133) v = trow[k - 128];
                a[2 * j + u] = (_Float16)v;
            }
        }
        return a;
    };
    hidden32(stateA, 5, P.ws + (size_t)OFF_STATE * FRAGE, P.state_b, &X[0][256], XSTR, lane);

    __syncthreads();

    // ---- zero card_logits for rows no stage covers (count == 4) ----
    float* outCard = P.out;
#pragma unroll
    for (int t = 0; t < 2; ++t) {
#pragma unroll
        for (int nt = 0; nt < 2; ++nt) {
#pragma unroll
            for (int r = 0; r < 8; ++r) {
                int row = t * 16 + r + hi * 8;
                if (Cnt[row] > 3)
                    outCard[(size_t)(rowBase + row) * 32 + nt * 16 + m] = 0.f;
            }
        }
    }

    // ---- dense stage branches + heads, masked select by count ----
    const int    kst[4]  = {16, 17, 17, 18};
    const size_t stOf[4] = {OFF_ST0, OFF_ST1, OFF_ST2, OFF_ST3};
    const float* stB[4]  = {P.stage_b0, P.stage_b1, P.stage_b2, P.stage_b3};
    const float* hdB[4]  = {P.head_b0,  P.head_b1,  P.head_b2,  P.head_b3};

    for (int s = 0; s < 4; ++s) {
        hidden32([&](int kt, int tile) {
                     return load_a_lds(&X[tile * 16 + m][0], kt, hi);
                 },
                 kst[s], P.ws + stOf[s] * FRAGE, stB[s], &Y[0][0], 256, lane);
        __syncthreads();
        v8f c00, c01, c10, c11;
        gemm2x2(&Y[0][0], 256, 8, P.ws + (size_t)(OFF_HEAD + s * 16) * FRAGE, lane,
                c00, c01, c10, c11);
        const float* hb = hdB[s];
#pragma unroll
        for (int r = 0; r < 8; ++r) {
            int row0 = r + hi * 8;
            if (Cnt[row0] == s) {
                outCard[(size_t)(rowBase + row0) * 32 + m]      = c00[r] + hb[m];
                outCard[(size_t)(rowBase + row0) * 32 + 16 + m] = c01[r] + hb[16 + m];
            }
            int row1 = 16 + r + hi * 8;
            if (Cnt[row1] == s) {
                outCard[(size_t)(rowBase + row1) * 32 + m]      = c10[r] + hb[m];
                outCard[(size_t)(rowBase + row1) * 32 + 16 + m] = c11[r] + hb[16 + m];
            }
        }
        __syncthreads();
    }

    // ---- combined bid(9)/announce(10)/value(1) heads from base ----
    {
        v8f c00, c01, c10, c11;
        gemm2x2(&X[0][0], XSTR, 16, P.ws + (size_t)OFF_COMB * FRAGE, lane,
                c00, c01, c10, c11);
        const int nB = P.nB;
        float* outBid = P.out + (size_t)nB * 32;
        float* outAnn = P.out + (size_t)nB * 41;
        float* outVal = P.out + (size_t)nB * 51;
#pragma unroll
        for (int t = 0; t < 2; ++t) {
            v8f& c0 = t ? c10 : c00;
            v8f& c1 = t ? c11 : c01;
#pragma unroll
            for (int r = 0; r < 8; ++r) {
                int row = rowBase + t * 16 + r + hi * 8;
                {   // nt=0 -> cols 0..15
                    int n = m; float v = c0[r];
                    if (n < 9) outBid[(size_t)row * 9 + n] = v + P.bid_b[n];
                    else       outAnn[(size_t)row * 10 + (n - 9)] = v + P.ann_b[n - 9];
                }
                {   // nt=1 -> cols 16..31
                    int n = 16 + m; float v = c1[r];
                    if (n < 19)       outAnn[(size_t)row * 10 + (n - 9)] = v + P.ann_b[n - 9];
                    else if (n == 19) outVal[row] = v + P.val_b[0];
                }
            }
        }
    }
}

extern "C" void kernel_launch(void* const* d_in, const int* in_sizes, int n_in,
                              void* d_out, int out_size, void* d_ws, size_t ws_size,
                              hipStream_t stream) {
    (void)n_in; (void)out_size;
    if (ws_size < (size_t)TOTAL_FRAGS * FRAGE * sizeof(_Float16)) return;

    const float* hand   = (const float*)d_in[0];
    const float* prob   = (const float*)d_in[1];
    const int*   tables = (const int*)  d_in[2];
    const float* trumps = (const float*)d_in[3];
    const float* emb    = (const float*)d_in[4];
    const float* hand_W = (const float*)d_in[5];
    const float* hand_b = (const float*)d_in[6];
    const float* state_W= (const float*)d_in[7];
    const float* state_b= (const float*)d_in[8];
    const float* stW[4] = {(const float*)d_in[9],  (const float*)d_in[10],
                           (const float*)d_in[11], (const float*)d_in[12]};
    const float* stB[4] = {(const float*)d_in[13], (const float*)d_in[14],
                           (const float*)d_in[15], (const float*)d_in[16]};
    const float* hdW[4] = {(const float*)d_in[17], (const float*)d_in[18],
                           (const float*)d_in[19], (const float*)d_in[20]};
    const float* hdB[4] = {(const float*)d_in[21], (const float*)d_in[22],
                           (const float*)d_in[23], (const float*)d_in[24]};
    const float* bid_W  = (const float*)d_in[25];
    const float* bid_b  = (const float*)d_in[26];
    const float* ann_W  = (const float*)d_in[27];
    const float* ann_b  = (const float*)d_in[28];
    const float* val_W  = (const float*)d_in[29];
    const float* val_b  = (const float*)d_in[30];

    _Float16* ws = (_Float16*)d_ws;
    const int nB = in_sizes[0] / 32;   // 65536

    PackParams pk = {hand_W, state_W, stW[0], stW[1], stW[2], stW[3],
                     hdW[0], hdW[1], hdW[2], hdW[3], bid_W, ann_W, val_W, ws};
    ppo_pack_weights<<<TOTAL_FRAGS, 256, 0, stream>>>(pk);

    NetParams np = {hand, prob, tables, trumps, emb,
                    hand_b, state_b,
                    stB[0], stB[1], stB[2], stB[3],
                    hdB[0], hdB[1], hdB[2], hdB[3],
                    bid_b, ann_b, val_b,
                    ws, (float*)d_out, nB};
    int blocks = nB / 32;              // 2048 blocks x 1 wave x 32 rows
    ppo_fused<<<blocks, 32, 0, stream>>>(np);
}
